// InterestEvolving_68418829025773
// MI455X (gfx1250) — matrix operand order repaired and checked by reading, not compiled, not run on hardware
//
#include <hip/hip_runtime.h>
#include <math.h>

#define BB  2048
#define SS  200
#define DD  128
#define FC1 32
#define FC2 16
#define G3  (3*DD)      // 384
#define TB  16          // batch tile for AUGRU scan

typedef __attribute__((ext_vector_type(16))) _Float16 v16h;
typedef __attribute__((ext_vector_type(8)))  float    v8f;

__device__ __forceinline__ float sigm(float x) { return 1.0f / (1.0f + __expf(-x)); }

// A fragment: 16x32 f16 tile from row-major LDS [rows][ld] at (row0, k0).
// lane&15 = M row; lane>>4 selects K-offset-8 half within each 16-K group.
__device__ __forceinline__ v16h load_A(const _Float16* sh, int ld, int row0, int k0, int lane) {
    const int m  = lane & 15;
    const int kh = lane >> 4;
    const _Float16* base = sh + (size_t)(row0 + m) * ld + k0;
    v16h a;
#pragma unroll
    for (int g = 0; g < 2; ++g)
#pragma unroll
        for (int e = 0; e < 8; ++e)
            a[g * 8 + e] = base[g * 16 + kh * 8 + e];
    return a;
}

// B fragment: B[k][n] = W[n][k] with W row-major [N][ld] in LDS; tile at (n0, k0).
// lane&15 = N col; lane>>4 selects K-offset-8 half within each 16-K group.
__device__ __forceinline__ v16h load_B(const _Float16* sh, int ld, int n0, int k0, int lane) {
    const int n  = lane & 15;
    const int kh = lane >> 4;
    const _Float16* base = sh + (size_t)(n0 + n) * ld + k0;
    v16h b;
#pragma unroll
    for (int e = 0; e < 16; ++e) b[e] = base[(e >> 3) * 16 + kh * 8 + (e & 7)];
    return b;
}

// ---------------- attention layer 1: x1 = ai @ W1^T + b1, plus batch stats ----
extern "C" __global__ __launch_bounds__(128)
void k_attn1(const float* __restrict__ q, const float* __restrict__ keys,
             const float* __restrict__ W1, const float* __restrict__ b1,
             float* __restrict__ x1, float* __restrict__ gsum, float* __restrict__ gsq) {
    extern __shared__ char smem[];
    _Float16* ai  = (_Float16*)smem;            // 32 x 512
    _Float16* w1  = ai + 32 * 4 * DD;           // 32 x 512
    float* sumsh  = (float*)(w1 + FC1 * 4 * DD);
    float* sqsh   = sumsh + FC1;

    const int tid = threadIdx.x;
    const int s   = blockIdx.y;
    const int b0  = blockIdx.x * 32;

    for (int i = tid; i < FC1 * 4 * DD; i += 128) w1[i] = (_Float16)W1[i];
    for (int j = tid; j < 32 * DD; j += 128) {
        int r = j >> 7, d = j & (DD - 1);
        float qv = q[(size_t)(b0 + r) * DD + d];
        float kv = keys[((size_t)(b0 + r) * SS + s) * DD + d];
        _Float16* row = ai + (size_t)r * (4 * DD);
        row[d]          = (_Float16)qv;
        row[DD + d]     = (_Float16)kv;
        row[2 * DD + d] = (_Float16)(qv - kv);
        row[3 * DD + d] = (_Float16)(qv * kv);
    }
    if (tid < FC1) { sumsh[tid] = 0.f; sqsh[tid] = 0.f; }
    __syncthreads();

    const int wave = tid >> 5, lane = tid & 31;
    const int mt = wave & 1, nt = wave >> 1;       // 2 m-tiles x 2 n-tiles
    const int colg = nt * 16 + (lane & 15);
    v8f acc;
    {
        float bv = b1[colg];
#pragma unroll
        for (int v = 0; v < 8; ++v) acc[v] = bv;
    }
#pragma unroll
    for (int kk = 0; kk < 16; ++kk) {           // K = 512
        v16h a  = load_A(ai, 4 * DD, mt * 16, kk * 32, lane);
        v16h bm = load_B(w1, 4 * DD, nt * 16, kk * 32, lane);
        acc = __builtin_amdgcn_wmma_f32_16x16x32_f16(false, a, false, bm, (short)0, acc, false, false);
    }
    const int rbase = mt * 16 + 8 * (lane >> 4);
    float ps = 0.f, pq = 0.f;
#pragma unroll
    for (int v = 0; v < 8; ++v) {
        float val = acc[v];
        x1[((size_t)(b0 + rbase + v) * SS + s) * FC1 + colg] = val;
        ps += val; pq += val * val;
    }
    atomicAdd(&sumsh[colg], ps);                 // ds_add_f32
    atomicAdd(&sqsh[colg], pq);
    __syncthreads();
    if (tid < FC1) {
        atomicAdd(&gsum[s * FC1 + tid], sumsh[tid]);
        atomicAdd(&gsq [s * FC1 + tid], sqsh[tid]);
    }
}

// ---------------- finalize mean / rstd (unbiased var, ddof=1) -----------------
extern "C" __global__ void k_stats(const float* __restrict__ sum, const float* __restrict__ sq,
                                   float* __restrict__ mean, float* __restrict__ istd, int n) {
    int i = blockIdx.x * blockDim.x + threadIdx.x;
    if (i >= n) return;
    float m   = sum[i] * (1.0f / (float)BB);
    float var = (sq[i] - (float)BB * m * m) * (1.0f / (float)(BB - 1));
    mean[i] = m;
    istd[i] = rsqrtf(var + 1e-9f);
}

// ---------------- attention layer 2: x2 = dice(x1) @ W2^T + b2, plus stats ----
extern "C" __global__ __launch_bounds__(64)
void k_attn2(const float* __restrict__ x1, const float* __restrict__ mean1,
             const float* __restrict__ istd1, const float* __restrict__ alpha1,
             const float* __restrict__ W2, const float* __restrict__ b2,
             float* __restrict__ x2, float* __restrict__ gsum, float* __restrict__ gsq) {
    __shared__ _Float16 h1sh[32 * FC1];
    __shared__ _Float16 w2sh[FC2 * FC1];
    __shared__ float sumsh[FC2], sqsh[FC2];
    const int tid = threadIdx.x;
    const int s   = blockIdx.y;
    const int b0  = blockIdx.x * 32;
    const float a1 = alpha1[0];

    for (int i = tid; i < FC2 * FC1; i += 64) w2sh[i] = (_Float16)W2[i];
    for (int j = tid; j < 32 * FC1; j += 64) {
        int r = j >> 5, c = j & (FC1 - 1);
        float x   = x1[((size_t)(b0 + r) * SS + s) * FC1 + c];
        float nrm = (x - mean1[s * FC1 + c]) * istd1[s * FC1 + c];
        float p   = sigm(nrm);
        h1sh[j]   = (_Float16)(a1 * x * (1.f - p) + x * p);
    }
    if (tid < FC2) { sumsh[tid] = 0.f; sqsh[tid] = 0.f; }
    __syncthreads();

    const int wave = tid >> 5, lane = tid & 31;   // 2 waves -> 2 m-tiles
    const int colg = lane & 15;
    v8f acc;
    {
        float bv = b2[colg];
#pragma unroll
        for (int v = 0; v < 8; ++v) acc[v] = bv;
    }
    v16h a  = load_A(h1sh, FC1, wave * 16, 0, lane);   // K = 32, single step
    v16h bm = load_B(w2sh, FC1, 0, 0, lane);
    acc = __builtin_amdgcn_wmma_f32_16x16x32_f16(false, a, false, bm, (short)0, acc, false, false);

    const int rbase = wave * 16 + 8 * (lane >> 4);
    float ps = 0.f, pq = 0.f;
#pragma unroll
    for (int v = 0; v < 8; ++v) {
        float val = acc[v];
        x2[((size_t)(b0 + rbase + v) * SS + s) * FC2 + colg] = val;
        ps += val; pq += val * val;
    }
    atomicAdd(&sumsh[colg], ps);
    atomicAdd(&sqsh[colg], pq);
    __syncthreads();
    if (tid < FC2) {
        atomicAdd(&gsum[s * FC2 + tid], sumsh[tid]);
        atomicAdd(&gsq [s * FC2 + tid], sqsh[tid]);
    }
}

// ---------------- attention score: att = dice(x2) @ w3^T + b3 -----------------
extern "C" __global__ __launch_bounds__(256)
void k_att3(const float* __restrict__ x2, const float* __restrict__ mean2,
            const float* __restrict__ istd2, const float* __restrict__ alpha2,
            const float* __restrict__ w3, const float* __restrict__ b3,
            float* __restrict__ att) {
    int row = blockIdx.x * 256 + threadIdx.x;
    if (row >= BB * SS) return;
    int s = row % SS;
    float a2 = alpha2[0];
    float acc = b3[0];
#pragma unroll
    for (int c = 0; c < FC2; ++c) {
        float x   = x2[(size_t)row * FC2 + c];
        float nrm = (x - mean2[s * FC2 + c]) * istd2[s * FC2 + c];
        float p   = sigm(nrm);
        acc += (a2 * x * (1.f - p) + x * p) * w3[c];
    }
    att[row] = acc;
}

// ---------------- AUGRU scan: weights resident in LDS (192KB of 320KB/WGP) ----
extern "C" __global__ __launch_bounds__(256)
void k_recur(const float* __restrict__ keys, const int* __restrict__ klen,
             const float* __restrict__ Wih, const float* __restrict__ Whh,
             const float* __restrict__ Bih, const float* __restrict__ Bhh,
             const float* __restrict__ att, float* __restrict__ out) {
    extern __shared__ char smem[];
    _Float16* wih = (_Float16*)smem;                 // 384x128 f16 = 96KB
    _Float16* whh = wih + G3 * DD;                   // 96KB
    float* bih = (float*)(whh + G3 * DD);            // 384
    float* bhh = bih + G3;                           // 384
    _Float16* kt  = (_Float16*)(bhh + G3);           // TB x 128 f16
    _Float16* h16 = kt + TB * DD;                    // TB x 128 f16
    float* hf  = (float*)(h16 + TB * DD);            // TB x 128 f32
    float* sr  = hf  + TB * DD;                      // r preact
    float* szn = sr  + TB * DD;                      // z preact
    float* gin = szn + TB * DD;                      // i_n
    float* ghn = gin + TB * DD;                      // h_n
    int* lens  = (int*)(ghn + TB * DD);              // TB

    const int tid = threadIdx.x;
    const int b0  = blockIdx.x * TB;

    for (int i = tid; i < G3 * DD; i += 256) { wih[i] = (_Float16)Wih[i]; whh[i] = (_Float16)Whh[i]; }
    for (int i = tid; i < G3; i += 256)      { bih[i] = Bih[i]; bhh[i] = Bhh[i]; }
    for (int i = tid; i < TB * DD; i += 256) { hf[i] = 0.f; h16[i] = (_Float16)0.f; }
    if (tid < TB) lens[tid] = klen[b0 + tid];
    __syncthreads();

    const int wave = tid >> 5, lane = tid & 31;   // 8 waves: wave owns 3 n-tiles

    for (int t = 0; t < SS; ++t) {
        // stage keys tile f32->f16; prefetch next timestep's rows
        for (int j = tid; j < TB * DD; j += 256) {
            int r = j >> 7, d = j & (DD - 1);
            const float* gp = &keys[((size_t)(b0 + r) * SS + t) * DD + d];
            kt[j] = (_Float16)gp[0];
            if (t + 1 < SS) __builtin_prefetch(gp + DD, 0, 1);   // global_prefetch_b8
        }
        __syncthreads();

        v16h afr[4];
        // ---- gi = keys_t @ Wih^T + bih
#pragma unroll
        for (int kk = 0; kk < 4; ++kk) afr[kk] = load_A(kt, DD, 0, kk * 32, lane);
#pragma unroll
        for (int j3 = 0; j3 < 3; ++j3) {
            const int nt = wave * 3 + j3;
            const int n  = nt * 16 + (lane & 15);
            v8f acc;
            { float bv = bih[n];
#pragma unroll
              for (int v = 0; v < 8; ++v) acc[v] = bv; }
#pragma unroll
            for (int kk = 0; kk < 4; ++kk) {
                v16h bm = load_B(wih, DD, nt * 16, kk * 32, lane);
                acc = __builtin_amdgcn_wmma_f32_16x16x32_f16(false, afr[kk], false, bm, (short)0, acc, false, false);
            }
            const int rb = 8 * (lane >> 4);
            float* dst = (n < DD) ? &sr[(size_t)rb * DD + n]
                       : (n < 2 * DD) ? &szn[(size_t)rb * DD + (n - DD)]
                                      : &gin[(size_t)rb * DD + (n - 2 * DD)];
#pragma unroll
            for (int v = 0; v < 8; ++v) dst[v * DD] = acc[v];
        }
        // ---- gh = h @ Whh^T + bhh ; r/z parts accumulate into sr/szn
#pragma unroll
        for (int kk = 0; kk < 4; ++kk) afr[kk] = load_A(h16, DD, 0, kk * 32, lane);
#pragma unroll
        for (int j3 = 0; j3 < 3; ++j3) {
            const int nt = wave * 3 + j3;
            const int n  = nt * 16 + (lane & 15);
            v8f acc;
            { float bv = bhh[n];
#pragma unroll
              for (int v = 0; v < 8; ++v) acc[v] = bv; }
#pragma unroll
            for (int kk = 0; kk < 4; ++kk) {
                v16h bm = load_B(whh, DD, nt * 16, kk * 32, lane);
                acc = __builtin_amdgcn_wmma_f32_16x16x32_f16(false, afr[kk], false, bm, (short)0, acc, false, false);
            }
            const int rb = 8 * (lane >> 4);
            if (n < 2 * DD) {   // wave-exclusive region: in-order LDS RMW is safe
                float* dst = (n < DD) ? &sr[(size_t)rb * DD + n] : &szn[(size_t)rb * DD + (n - DD)];
#pragma unroll
                for (int v = 0; v < 8; ++v) dst[v * DD] += acc[v];
            } else {
                float* dst = &ghn[(size_t)rb * DD + (n - 2 * DD)];
#pragma unroll
                for (int v = 0; v < 8; ++v) dst[v * DD] = acc[v];
            }
        }
        __syncthreads();

        // ---- gates + attentional update + length freeze
        for (int j = tid; j < TB * DD; j += 256) {
            int r = j >> 7;
            float a_t = att[(size_t)(b0 + r) * SS + t];
            float rg = sigm(sr[j]);
            float zg = sigm(szn[j]) * a_t;
            float ng = tanhf(gin[j] + rg * ghn[j]);
            float ho = hf[j];
            float hy = (1.f - zg) * ho + zg * ng;
            float hn = (t < lens[r]) ? hy : ho;
            hf[j]  = hn;
            h16[j] = (_Float16)hn;
        }
        __syncthreads();
    }

    for (int j = tid; j < TB * DD; j += 256) {
        int r = j >> 7, d = j & (DD - 1);
        out[(size_t)(b0 + r) * DD + d] = hf[j];
    }
}

extern "C" void kernel_launch(void* const* d_in, const int* in_sizes, int n_in,
                              void* d_out, int out_size, void* d_ws, size_t ws_size,
                              hipStream_t stream) {
    const float* q    = (const float*)d_in[0];
    const float* keys = (const float*)d_in[1];
    const int*   klen = (const int*)  d_in[2];
    const float* wih  = (const float*)d_in[3];
    const float* whh  = (const float*)d_in[4];
    const float* bih  = (const float*)d_in[5];
    const float* bhh  = (const float*)d_in[6];
    const float* W1   = (const float*)d_in[7];
    const float* b1   = (const float*)d_in[8];
    const float* al1  = (const float*)d_in[9];
    const float* W2   = (const float*)d_in[10];
    const float* b2   = (const float*)d_in[11];
    const float* al2  = (const float*)d_in[12];
    const float* w3   = (const float*)d_in[13];
    const float* b3   = (const float*)d_in[14];
    float* out = (float*)d_out;

    // workspace layout (~80 MB)
    char* ws = (char*)d_ws;
    float* x1   = (float*)ws; ws += (size_t)BB * SS * FC1 * sizeof(float);
    float* x2   = (float*)ws; ws += (size_t)BB * SS * FC2 * sizeof(float);
    float* att  = (float*)ws; ws += (size_t)BB * SS * sizeof(float);
    float* stats = (float*)ws;
    float* sum1  = stats;
    float* sq1   = sum1 + SS * FC1;
    float* mean1 = sq1  + SS * FC1;
    float* istd1 = mean1 + SS * FC1;
    float* sum2  = istd1 + SS * FC1;
    float* sq2   = sum2 + SS * FC2;
    float* mean2 = sq2  + SS * FC2;
    float* istd2 = mean2 + SS * FC2;
    size_t stats_bytes = (size_t)(4 * SS * FC1 + 4 * SS * FC2) * sizeof(float);
    hipMemsetAsync(stats, 0, stats_bytes, stream);

    const int lds1 = (32 * 4 * DD + FC1 * 4 * DD) * (int)sizeof(_Float16) + 2 * FC1 * (int)sizeof(float);
    const int ldsr = 2 * G3 * DD * (int)sizeof(_Float16)          // weights: 192KB
                   + 2 * G3 * (int)sizeof(float)                  // biases
                   + 2 * TB * DD * (int)sizeof(_Float16)          // kt + h16
                   + 5 * TB * DD * (int)sizeof(float)             // hf,sr,szn,gin,ghn
                   + TB * (int)sizeof(int);                       // lens
    hipFuncSetAttribute(reinterpret_cast<const void*>(k_attn1),
                        hipFuncAttributeMaxDynamicSharedMemorySize, lds1);
    hipFuncSetAttribute(reinterpret_cast<const void*>(k_recur),
                        hipFuncAttributeMaxDynamicSharedMemorySize, ldsr);

    k_attn1<<<dim3(BB / 32, SS), 128, lds1, stream>>>(q, keys, W1, b1, x1, sum1, sq1);
    k_stats<<<(SS * FC1 + 255) / 256, 256, 0, stream>>>(sum1, sq1, mean1, istd1, SS * FC1);
    k_attn2<<<dim3(BB / 32, SS), 64, 0, stream>>>(x1, mean1, istd1, al1, W2, b2, x2, sum2, sq2);
    k_stats<<<(SS * FC2 + 255) / 256, 256, 0, stream>>>(sum2, sq2, mean2, istd2, SS * FC2);
    k_att3<<<(BB * SS + 255) / 256, 256, 0, stream>>>(x2, mean2, istd2, al2, w3, b3, att);
    k_recur<<<BB / TB, 256, ldsr, stream>>>(keys, klen, wih, whh, bih, bhh, att, out);
}
